// GobalAttention_21827023798426
// MI455X (gfx1250) — matrix operand order repaired and checked by reading, not compiled, not run on hardware
//
#include <hip/hip_runtime.h>

typedef __attribute__((ext_vector_type(16))) _Float16 v16h;
typedef __attribute__((ext_vector_type(2)))  _Float16 h2;
typedef __attribute__((ext_vector_type(8)))  float    v8f;

#define WMMA16(A, Bf, C) \
  __builtin_amdgcn_wmma_f32_16x16x32_f16(false, (A), false, (Bf), (short)0, (C), false, false)

#define D_HEAD 64
#define NTOP 9
#define KSTR 72   /* LDS row stride (halves) for K tile: 144B, 16B-aligned rows */
#define SSTR 36   /* LDS row stride (floats) for score tile: 144B */

// Branchless sorted (descending) top-9 insertion: cmp + cndmask chain.
__device__ __forceinline__ void topk_insert(float (&val)[NTOP], int (&idx)[NTOP],
                                            float cv, int ci) {
#pragma unroll
  for (int i = 0; i < NTOP; ++i) {
    bool gt = cv > val[i];
    float tvv = val[i];
    int   tii = idx[i];
    val[i] = gt ? cv : tvv;
    idx[i] = gt ? ci : tii;
    cv = gt ? tvv : cv;
    ci = gt ? tii : ci;
  }
}

// Transpose V: (B, d, N) -> (B, N, d) so the final gather is contiguous.
__global__ __launch_bounds__(256) void vtrans_k(const float* __restrict__ v,
                                                float* __restrict__ vt,
                                                int d, int N) {
  __shared__ float tile[32][33];
  const int b  = blockIdx.z;
  const int c0 = blockIdx.y * 32;
  const int n0 = blockIdx.x * 32;
  const int tx = threadIdx.x, ty = threadIdx.y;  // (32, 8)
  const float* vb  = v  + (size_t)b * d * N;
  float*       vtb = vt + (size_t)b * N * d;
#pragma unroll
  for (int i = 0; i < 32; i += 8)
    tile[ty + i][tx] = vb[(size_t)(c0 + ty + i) * N + (n0 + tx)];
  __syncthreads();
#pragma unroll
  for (int i = 0; i < 32; i += 8)
    vtb[(size_t)(n0 + ty + i) * d + (c0 + tx)] = tile[tx][ty + i];
}

template <bool VT>
__global__ __launch_bounds__(128) void topk_attn_k(const float* __restrict__ q,
                                                   const float* __restrict__ k,
                                                   const float* __restrict__ vsel,
                                                   float* __restrict__ out,
                                                   int N) {
  // K tile staged column-major: kh[col][K] -> fragments are contiguous 32B runs.
  __shared__ __align__(16) _Float16 kh[32][KSTR];
  __shared__ __align__(16) _Float16 kl[32][KSTR];
  __shared__ __align__(16) float sc[4][16][SSTR];
  __shared__ float mval[4][32][NTOP];
  __shared__ int   midx[4][32][NTOP];
  __shared__ float wgt[4][16][NTOP];
  __shared__ int   gsel[4][16][NTOP];

  const int tilesPerBatch = N / 64;
  const int b    = blockIdx.x / tilesPerBatch;
  const int n0b  = (blockIdx.x % tilesPerBatch) * 64;
  const int wave = threadIdx.x >> 5;
  const int lane = threadIdx.x & 31;
  const int lo16 = lane & 15;
  const int hi   = lane >> 4;              // which 16-lane half
  const int n0   = n0b + wave * 16;        // this wave's 16 query rows

  const float* qb = q + (size_t)b * D_HEAD * N;
  const float* kb = k + (size_t)b * D_HEAD * N;

  // ---- A fragments (Q rows), split f32 -> f16 hi/lo; resident for the sweep.
  // A k order used here (contiguous, same as B): lane M = lane&15; elem e -> K = e + ko,
  // ko = 16*(lane>=16).
  v16h a_hi0, a_lo0, a_hi1, a_lo1;
#pragma unroll
  for (int e = 0; e < 16; ++e) {
    const int K0 = e + hi * 16;
    const float x0 = qb[(size_t)K0 * N + (n0 + lo16)];
    const _Float16 h0 = (_Float16)x0;
    a_hi0[e] = h0;
    a_lo0[e] = (_Float16)(x0 - (float)h0);
    const float x1 = qb[(size_t)(K0 + 32) * N + (n0 + lo16)];
    const _Float16 h1 = (_Float16)x1;
    a_hi1[e] = h1;
    a_lo1[e] = (_Float16)(x1 - (float)h1);
  }

  float tv[NTOP];
  int   ti[NTOP];
#pragma unroll
  for (int t = 0; t < NTOP; ++t) { tv[t] = -3.402823466e38f; ti[t] = 0; }

  // ---- sweep over key columns, 32 at a time
  for (int ct = 0; ct < N; ct += 32) {
    // cooperative stage: thread handles channel pair (c, c+1) of column j
#pragma unroll
    for (int i = 0; i < 8; ++i) {
      const int p = threadIdx.x + 128 * i;     // 0..1023 pairs
      const int j = p & 31;
      const int c = (p >> 5) * 2;
      const float x0 = kb[(size_t)c * N + (ct + j)];
      const float x1 = kb[(size_t)(c + 1) * N + (ct + j)];
      const _Float16 h0 = (_Float16)x0;
      const _Float16 h1 = (_Float16)x1;
      const _Float16 l0 = (_Float16)(x0 - (float)h0);
      const _Float16 l1 = (_Float16)(x1 - (float)h1);
      *reinterpret_cast<h2*>(&kh[j][c]) = (h2){h0, h1};
      *reinterpret_cast<h2*>(&kl[j][c]) = (h2){l0, l1};
    }
    __syncthreads();

    // B fragments: col = lane&15 (+16 for 2nd tile), elem e -> K = e + 16*hi
    // (+32 for 2nd K-half). Contiguous 32B in LDS -> 2x ds_load_b128 each.
    v16h bh0a, bl0a, bh1a, bl1a, bh0b, bl0b, bh1b, bl1b;
    {
      const int kofs = 16 * hi;
      __builtin_memcpy(&bh0a, &kh[lo16][kofs],            32);
      __builtin_memcpy(&bl0a, &kl[lo16][kofs],            32);
      __builtin_memcpy(&bh1a, &kh[lo16][kofs + 32],       32);
      __builtin_memcpy(&bl1a, &kl[lo16][kofs + 32],       32);
      __builtin_memcpy(&bh0b, &kh[16 + lo16][kofs],       32);
      __builtin_memcpy(&bl0b, &kl[16 + lo16][kofs],       32);
      __builtin_memcpy(&bh1b, &kh[16 + lo16][kofs + 32],  32);
      __builtin_memcpy(&bl1b, &kl[16 + lo16][kofs + 32],  32);
    }
    __syncthreads();   // all waves done reading -> safe to restage next iter

    // split-precision product: hi*hi + hi*lo + lo*hi (lo*lo negligible)
    v8f c0 = {};
    c0 = WMMA16(a_hi0, bh0a, c0);
    c0 = WMMA16(a_hi1, bh1a, c0);
    c0 = WMMA16(a_hi0, bl0a, c0);
    c0 = WMMA16(a_hi1, bl1a, c0);
    c0 = WMMA16(a_lo0, bh0a, c0);
    c0 = WMMA16(a_lo1, bh1a, c0);
    v8f c1 = {};
    c1 = WMMA16(a_hi0, bh0b, c1);
    c1 = WMMA16(a_hi1, bh1b, c1);
    c1 = WMMA16(a_hi0, bl0b, c1);
    c1 = WMMA16(a_hi1, bl1b, c1);
    c1 = WMMA16(a_lo0, bh0b, c1);
    c1 = WMMA16(a_lo1, bh1b, c1);

    // C layout: vgpr r, lane -> row r + 8*hi, col lane&15; stage to LDS
#pragma unroll
    for (int r = 0; r < 8; ++r) {
      sc[wave][r + 8 * hi][lo16]      = c0[r];
      sc[wave][r + 8 * hi][16 + lo16] = c1[r];
    }
    __builtin_amdgcn_wave_barrier();   // keep store->load order; DS is in-order per wave

    // top-9 scan: 2 lanes per row, 16 candidates each, vector-loaded
    {
      const int row = lo16;
      const int cb  = 16 * hi;
      const float4* srow = reinterpret_cast<const float4*>(&sc[wave][row][cb]);
      const float4 s0 = srow[0], s1 = srow[1], s2 = srow[2], s3 = srow[3];
      const float cand[16] = {s0.x, s0.y, s0.z, s0.w, s1.x, s1.y, s1.z, s1.w,
                              s2.x, s2.y, s2.z, s2.w, s3.x, s3.y, s3.z, s3.w};
#pragma unroll
      for (int j = 0; j < 16; ++j) {
        const float cv = cand[j];
        // wave-uniform gate: chain only if some lane beats its current 9th
        if (__ballot(cv > tv[NTOP - 1]) != 0ull)
          topk_insert(tv, ti, cv, ct + cb + j);
      }
    }
  }

  // ---- merge the two per-row lists, softmax, broadcast
#pragma unroll
  for (int t = 0; t < NTOP; ++t) { mval[wave][lane][t] = tv[t]; midx[wave][lane][t] = ti[t]; }
  __syncthreads();

  if (lane < 16) {
#pragma unroll
    for (int t = 0; t < NTOP; ++t)
      topk_insert(tv, ti, mval[wave][lane + 16][t], midx[wave][lane + 16][t]);
    const float mx = tv[0];                 // sorted descending
    float w[NTOP], s = 0.f;
#pragma unroll
    for (int t = 0; t < NTOP; ++t) {        // reference: attn = dot / SCALE = dot * 8
      w[t] = __expf((tv[t] - mx) * 8.0f);
      s += w[t];
    }
    const float inv = 1.0f / s;
#pragma unroll
    for (int t = 0; t < NTOP; ++t) { wgt[wave][lane][t] = w[t] * inv; gsel[wave][lane][t] = ti[t]; }
  }
  __syncthreads();

  // ---- weighted gather of V rows: 2 lanes/row, 32 channels each
  const size_t rstr = VT ? (size_t)D_HEAD : (size_t)1;
  const size_t cstr = VT ? (size_t)1 : (size_t)N;
  const float* vb = vsel + (size_t)b * D_HEAD * N;
  const int row = lo16;
  const int cb  = hi * 32;
  float acc[32];
#pragma unroll
  for (int c = 0; c < 32; ++c) acc[c] = 0.f;
#pragma unroll
  for (int t = 0; t < NTOP; ++t) {
    const float wv = wgt[wave][row][t];
    const int   id = gsel[wave][row][t];
    const float* src = vb + (size_t)id * rstr + (size_t)cb * cstr;
#pragma unroll
    for (int c = 0; c < 32; ++c) acc[c] += wv * src[(size_t)c * cstr];
  }
  float* ob = out + ((size_t)b * N + (n0 + row)) * D_HEAD + cb;
#pragma unroll
  for (int c = 0; c < 32; ++c) ob[c] = acc[c];
}

extern "C" void kernel_launch(void* const* d_in, const int* in_sizes, int n_in,
                              void* d_out, int out_size, void* d_ws, size_t ws_size,
                              hipStream_t stream) {
  (void)n_in; (void)out_size;
  const float* q = (const float*)d_in[0];
  const float* k = (const float*)d_in[1];
  const float* v = (const float*)d_in[2];
  float* out = (float*)d_out;

  const int N = 96 * 96;
  const int B = in_sizes[0] / (D_HEAD * N);       // = 4
  const size_t vt_bytes = (size_t)B * N * D_HEAD * sizeof(float);

  const dim3 grid((unsigned)(B * (N / 64)));
  const dim3 blk(128);

  if (ws_size >= vt_bytes) {
    float* vt = (float*)d_ws;
    vtrans_k<<<dim3(N / 32, D_HEAD / 32, B), dim3(32, 8), 0, stream>>>(v, vt, D_HEAD, N);
    topk_attn_k<true><<<grid, blk, 0, stream>>>(q, k, vt, out, N);
  } else {
    topk_attn_k<false><<<grid, blk, 0, stream>>>(q, k, v, out, N);
  }
}
